// SelfAttention_13383118095074
// MI455X (gfx1250) — matrix operand order repaired and checked by reading, not compiled, not run on hardware
//
#include <hip/hip_runtime.h>

typedef __attribute__((ext_vector_type(16))) _Float16 v16h;
typedef __attribute__((ext_vector_type(8)))  _Float16 v8h;
typedef __attribute__((ext_vector_type(4)))  _Float16 v4h;
typedef __attribute__((ext_vector_type(8)))  float    v8f;

#define B_   2
#define L_   2048
#define D_   2048
#define H_   32
#define KV_  8
#define HD_  64
#define REP_ 4

// ---------------------------------------------------------------------------
// CDNA5 async global->LDS copy (VGLOBAL encoding, tracked by ASYNCcnt):
//   global_load_async_to_lds_b128 vDST(lds addr), v[ADDR](64b global), off
// LDS offset = low 32 bits of the generic shared-aperture address.
// ---------------------------------------------------------------------------
__device__ __forceinline__ uint32_t lds_off(const void* p) {
  return (uint32_t)(uintptr_t)p;
}
__device__ __forceinline__ void async_b128(uint32_t lds, const void* g) {
  asm volatile("global_load_async_to_lds_b128 %0, %1, off"
               :: "v"(lds), "v"(g) : "memory");
}
__device__ __forceinline__ void wait_async0() {
  asm volatile("s_wait_asynccnt 0x0" ::: "memory");
}

// ---------------------------------------------------------------------------
// Fragment loaders (CDNA5 WMMA 16x16x32 f16, wave32)
// A 16x32: lane holds row (lane&15); chunks k = kb..kb+7 and kb+16..kb+23,
//          kb = (lane>>4)*8  -> two 16B loads, 16 halfs apart.
// B 32x16: lane holds col (lane&15); same k chunks as A (kb..kb+7, kb+16..kb+23).
// C/D 16x16 f32: vgpr i, lanes 0-15 -> (M=i, N=lane); lanes 16-31 -> (M=i+8).
// ---------------------------------------------------------------------------
__device__ __forceinline__ v16h ld_frag_a(const _Float16* p) {
  v8h lo = *(const v8h*)p;
  v8h hi = *(const v8h*)(p + 16);
  return __builtin_shufflevector(lo, hi, 0,1,2,3,4,5,6,7,8,9,10,11,12,13,14,15);
}
__device__ __forceinline__ v16h ld_frag_b(const _Float16* p) {
  v8h lo = *(const v8h*)p;
  v8h hi = *(const v8h*)(p + 16);
  return __builtin_shufflevector(lo, hi, 0,1,2,3,4,5,6,7,8,9,10,11,12,13,14,15);
}

// ---------------------------------------------------------------------------
// fp32 -> fp16 bulk convert (vectorized x4)
// ---------------------------------------------------------------------------
__global__ void k_cvt4(const float* __restrict__ s, _Float16* __restrict__ d,
                       long n4) {
  long i = blockIdx.x * (long)blockDim.x + threadIdx.x;
  if (i >= n4) return;
  float4 f = ((const float4*)s)[i];
  v4h o = {(_Float16)f.x, (_Float16)f.y, (_Float16)f.z, (_Float16)f.w};
  ((v4h*)d)[i] = o;
}

// ---------------------------------------------------------------------------
// W[K][N] fp32 -> Wt[N][K] fp16, 32x32 LDS tiles. grid=(K/32, N/32), block=(32,8)
// ---------------------------------------------------------------------------
__global__ void k_wtrans(const float* __restrict__ W, _Float16* __restrict__ Wt,
                         int K, int N) {
  __shared__ _Float16 t[32][33];
  const int k0 = blockIdx.x * 32, n0 = blockIdx.y * 32;
  const int x = threadIdx.x;
  #pragma unroll
  for (int yy = 0; yy < 4; ++yy) {
    int y = threadIdx.y * 4 + yy;
    t[y][x] = (_Float16)W[(size_t)(k0 + y) * N + n0 + x];
  }
  __syncthreads();
  #pragma unroll
  for (int yy = 0; yy < 4; ++yy) {
    int y = threadIdx.y * 4 + yy;
    Wt[(size_t)(n0 + y) * K + k0 + x] = t[x][y];
  }
}

// ---------------------------------------------------------------------------
// V fp32 [b][l][g][hd] -> Vt fp16 [b][g][hd][l]. grid=(L/32, HD/32, B*KV), block=(32,8)
// ---------------------------------------------------------------------------
__global__ void k_vtrans(const float* __restrict__ V, _Float16* __restrict__ Vt) {
  __shared__ _Float16 t[32][33];
  const int l0 = blockIdx.x * 32, hd0 = blockIdx.y * 32;
  const int b = blockIdx.z / KV_, g = blockIdx.z % KV_;
  const int x = threadIdx.x;
  #pragma unroll
  for (int yy = 0; yy < 4; ++yy) {
    int y = threadIdx.y * 4 + yy;  // l offset
    t[y][x] = (_Float16)V[(((size_t)b * L_ + l0 + y) * KV_ + g) * HD_ + hd0 + x];
  }
  __syncthreads();
  #pragma unroll
  for (int yy = 0; yy < 4; ++yy) {
    int y = threadIdx.y * 4 + yy;  // hd offset
    Vt[(((size_t)b * KV_ + g) * HD_ + hd0 + y) * L_ + l0 + x] = t[x][y];
  }
}

// ---------------------------------------------------------------------------
// RoPE: src fp32 [B*L][NH*HD] -> dst fp16 [B][NH][L][HD] (head-major)
// ---------------------------------------------------------------------------
__global__ void k_rope(const float* __restrict__ src,
                       const float* __restrict__ cosT,
                       const float* __restrict__ sinT,
                       _Float16* __restrict__ dst, int NH, long total) {
  long i = blockIdx.x * (long)blockDim.x + threadIdx.x;
  if (i >= total) return;
  const int HD2 = HD_ / 2;
  int d2 = (int)(i % HD2);
  long t = i / HD2;
  int hh = (int)(t % NH);  t /= NH;
  int l  = (int)(t % L_);
  int b  = (int)(t / L_);
  size_t si = ((size_t)(b * L_ + l) * NH + hh) * HD_ + 2 * d2;
  float a  = src[si];
  float bb = src[si + 1];
  float c = cosT[(size_t)l * HD2 + d2];
  float s = sinT[(size_t)l * HD2 + d2];
  size_t o = (((size_t)b * NH + hh) * L_ + l) * HD_ + 2 * d2;
  dst[o]     = (_Float16)(a * c - bb * s);
  dst[o + 1] = (_Float16)(a * s + bb * c);
}

// ---------------------------------------------------------------------------
// GEMM: C[M][N] = A[M][K] * B, A fp16 [M][K], Bt fp16 = B^T [N][K], C fp32.
// Tiles 128x128x64, 256 threads = 8 waves (2x4), wave tile 64x32.
// Double-buffered LDS fed by async global->LDS b128; ONE barrier per K tile.
// ---------------------------------------------------------------------------
#define GBM 128
#define GBN 128
#define GBK 64
#define GLDK 72  // padded LDS row stride in halfs

__global__ void __launch_bounds__(256)
k_gemm(const _Float16* __restrict__ A, const _Float16* __restrict__ Bt,
       float* __restrict__ C, int M, int N, int K) {
  __shared__ _Float16 As[2][GBM * GLDK];
  __shared__ _Float16 Bs[2][GBN * GLDK];
  const int tid  = threadIdx.x;
  const int wave = tid >> 5, lane = tid & 31;
  const int wm = wave >> 2, wn = wave & 3;
  const int lh = lane & 15, ls = lane >> 4;
  const int m0 = blockIdx.y * GBM;
  const int n0 = blockIdx.x * GBN;

  // Per-thread staging slots: 4 x 16B each for A and B per K tile.
  int rr[4], cc[4];
  #pragma unroll
  for (int p = 0; p < 4; ++p) {
    int idx = (p * 256 + tid) * 8;
    rr[p] = idx >> 6;
    cc[p] = idx & 63;
  }

  auto issue = [&](int buf, int k0) {
    #pragma unroll
    for (int p = 0; p < 4; ++p) {
      async_b128(lds_off(&As[buf][rr[p] * GLDK + cc[p]]),
                 &A[(size_t)(m0 + rr[p]) * K + k0 + cc[p]]);
      async_b128(lds_off(&Bs[buf][rr[p] * GLDK + cc[p]]),
                 &Bt[(size_t)(n0 + rr[p]) * K + k0 + cc[p]]);
    }
  };

  v8f acc[4][2];
  #pragma unroll
  for (int i = 0; i < 4; ++i)
    #pragma unroll
    for (int j = 0; j < 2; ++j) { v8f z = {}; acc[i][j] = z; }

  issue(0, 0);
  const int nt = K / GBK;
  for (int t = 0; t < nt; ++t) {
    wait_async0();       // drain this wave's async copies for tile t
    __syncthreads();     // all waves' pieces of tile t are in LDS
    if (t + 1 < nt) issue((t + 1) & 1, (t + 1) * GBK);  // stream next tile

    const _Float16* as = As[t & 1];
    const _Float16* bs = Bs[t & 1];
    #pragma unroll
    for (int ks = 0; ks < 2; ++ks) {
      v16h af[4], bf[2];
      #pragma unroll
      for (int i = 0; i < 4; ++i)
        af[i] = ld_frag_a(&as[(wm * 64 + i * 16 + lh) * GLDK + ks * 32 + ls * 8]);
      #pragma unroll
      for (int j = 0; j < 2; ++j)
        bf[j] = ld_frag_b(&bs[(wn * 32 + j * 16 + lh) * GLDK + ks * 32 + ls * 8]);
      #pragma unroll
      for (int i = 0; i < 4; ++i)
        #pragma unroll
        for (int j = 0; j < 2; ++j)
          acc[i][j] = __builtin_amdgcn_wmma_f32_16x16x32_f16(
              false, af[i], false, bf[j], (short)0, acc[i][j], false, false);
    }
  }

  #pragma unroll
  for (int i = 0; i < 4; ++i)
    #pragma unroll
    for (int j = 0; j < 2; ++j) {
      int row = m0 + wm * 64 + i * 16 + ls * 8;
      int col = n0 + wn * 32 + j * 16 + lh;
      #pragma unroll
      for (int r = 0; r < 8; ++r)
        C[(size_t)(row + r) * N + col] = acc[i][j][r];
    }
}

// ---------------------------------------------------------------------------
// Flash attention. grid = (L/64, B*H), block = 128 (4 waves).
// Double-buffered K/V tiles via async global->LDS; one barrier per KV tile.
// Wave w owns 16 query rows: S = Q K^T (8 WMMA), online softmax
// (shfl_xor within 16-lane halves), P re-fragmented via per-wave LDS,
// O += P V (8 WMMA).
// ---------------------------------------------------------------------------
#define FLDK 72

__global__ void __launch_bounds__(128)
k_flash(const _Float16* __restrict__ Qh, const _Float16* __restrict__ Kh,
        const _Float16* __restrict__ Vth, _Float16* __restrict__ AO) {
  const int qt = blockIdx.x;
  const int bh = blockIdx.y;
  const int b = bh / H_, h = bh % H_;
  const int g = h / REP_;
  const int tid = threadIdx.x;
  const int wave = tid >> 5, lane = tid & 31;
  const int lh = lane & 15, ls = lane >> 4;
  const int q0 = qt * 64 + wave * 16;

  const _Float16* Qb = Qh  + ((size_t)(b * H_  + h) * L_)  * HD_;
  const _Float16* Kb = Kh  + ((size_t)(b * KV_ + g) * L_)  * HD_;
  const _Float16* Vb = Vth + ((size_t)(b * KV_ + g) * HD_) * L_;

  __shared__ _Float16 Ks[2][64 * FLDK];
  __shared__ _Float16 Vs[2][64 * FLDK];
  __shared__ _Float16 Ps[4][16 * FLDK];

  int rr[4], cc[4];
  #pragma unroll
  for (int p = 0; p < 4; ++p) {
    int idx = (p * 128 + tid) * 8;
    rr[p] = idx >> 6;
    cc[p] = idx & 63;
  }
  auto issue = [&](int buf, int kv0) {
    #pragma unroll
    for (int p = 0; p < 4; ++p) {
      async_b128(lds_off(&Ks[buf][rr[p] * FLDK + cc[p]]),
                 &Kb[(size_t)(kv0 + rr[p]) * HD_ + cc[p]]);
      async_b128(lds_off(&Vs[buf][rr[p] * FLDK + cc[p]]),
                 &Vb[(size_t)rr[p] * L_ + kv0 + cc[p]]);
    }
  };

  v16h qf[2];
  #pragma unroll
  for (int ks = 0; ks < 2; ++ks)
    qf[ks] = ld_frag_a(Qb + (size_t)(q0 + lh) * HD_ + ks * 32 + ls * 8);

  float rmax[8], rsum[8];
  v8f of[4];
  #pragma unroll
  for (int r = 0; r < 8; ++r) { rmax[r] = -3.0e38f; rsum[r] = 0.0f; }
  #pragma unroll
  for (int n = 0; n < 4; ++n) { v8f z = {}; of[n] = z; }

  const float scale = 0.125f;  // 1/sqrt(HD)

  issue(0, 0);
  for (int j = 0; j <= qt; ++j) {
    wait_async0();
    __syncthreads();
    if (j + 1 <= qt) issue((j + 1) & 1, (j + 1) * 64);

    const _Float16* ks_ = Ks[j & 1];
    const _Float16* vs_ = Vs[j & 1];

    // S = Q K^T for this wave's 16 rows x 64 kv columns.
    v8f sf[4];
    #pragma unroll
    for (int n = 0; n < 4; ++n) {
      v8f z = {};
      sf[n] = z;
      #pragma unroll
      for (int ks = 0; ks < 2; ++ks) {
        v16h kb = ld_frag_b(&ks_[(n * 16 + lh) * FLDK + ks * 32 + ls * 8]);
        sf[n] = __builtin_amdgcn_wmma_f32_16x16x32_f16(
            false, qf[ks], false, kb, (short)0, sf[n], false, false);
      }
    }

    // Scale + causal mask (only the diagonal tile needs masking).
    if (j == qt) {
      #pragma unroll
      for (int n = 0; n < 4; ++n)
        #pragma unroll
        for (int r = 0; r < 8; ++r) {
          int qrel = wave * 16 + r + 8 * ls;
          int krel = n * 16 + lh;
          float v = sf[n][r] * scale;
          sf[n][r] = (krel > qrel) ? -3.0e38f : v;
        }
    } else {
      #pragma unroll
      for (int n = 0; n < 4; ++n)
        #pragma unroll
        for (int r = 0; r < 8; ++r)
          sf[n][r] *= scale;
    }

    // Row max (over 4 n-frags, then 16 lanes within each half).
    float mnew[8], alpha[8];
    #pragma unroll
    for (int r = 0; r < 8; ++r) {
      float t = fmaxf(fmaxf(sf[0][r], sf[1][r]), fmaxf(sf[2][r], sf[3][r]));
      #pragma unroll
      for (int m = 8; m >= 1; m >>= 1)
        t = fmaxf(t, __shfl_xor(t, m, 32));
      mnew[r]  = fmaxf(rmax[r], t);
      alpha[r] = __expf(rmax[r] - mnew[r]);
      rmax[r]  = mnew[r];
    }

    // P = exp(S - mnew): f16 into per-wave LDS; accumulate row sums.
    float psum[8];
    #pragma unroll
    for (int r = 0; r < 8; ++r) psum[r] = 0.0f;
    #pragma unroll
    for (int n = 0; n < 4; ++n)
      #pragma unroll
      for (int r = 0; r < 8; ++r) {
        float p = __expf(sf[n][r] - mnew[r]);
        psum[r] += p;
        Ps[wave][(r + 8 * ls) * FLDK + n * 16 + lh] = (_Float16)p;
      }
    #pragma unroll
    for (int r = 0; r < 8; ++r) {
      float t = psum[r];
      #pragma unroll
      for (int m = 8; m >= 1; m >>= 1)
        t += __shfl_xor(t, m, 32);
      rsum[r] = rsum[r] * alpha[r] + t;
      #pragma unroll
      for (int n = 0; n < 4; ++n)
        of[n][r] *= alpha[r];
    }

    // O += P @ V  (intra-wave LDS RAW ordered by DScnt).
    #pragma unroll
    for (int ks = 0; ks < 2; ++ks) {
      v16h pf = ld_frag_a(&Ps[wave][lh * FLDK + ks * 32 + ls * 8]);
      #pragma unroll
      for (int n = 0; n < 4; ++n) {
        v16h vb = ld_frag_b(&vs_[(n * 16 + lh) * FLDK + ks * 32 + ls * 8]);
        of[n] = __builtin_amdgcn_wmma_f32_16x16x32_f16(
            false, pf, false, vb, (short)0, of[n], false, false);
      }
    }
  }

  // Normalize and write attention output f16 [b][l][h*hd].
  #pragma unroll
  for (int r = 0; r < 8; ++r) {
    float inv = 1.0f / rsum[r];
    int qpos = q0 + r + 8 * ls;
    #pragma unroll
    for (int n = 0; n < 4; ++n) {
      size_t o = (size_t)(b * L_ + qpos) * (H_ * HD_) + h * HD_ + n * 16 + lh;
      AO[o] = (_Float16)(of[n][r] * inv);
    }
  }
}

// ---------------------------------------------------------------------------
// Host-side launch
// ---------------------------------------------------------------------------
extern "C" void kernel_launch(void* const* d_in, const int* in_sizes, int n_in,
                              void* d_out, int out_size, void* d_ws, size_t ws_size,
                              hipStream_t stream) {
  (void)in_sizes; (void)n_in; (void)out_size; (void)ws_size;

  const float* x  = (const float*)d_in[0];
  // d_in[1] = start_pos (0), d_in[4] = mask (causal, computed inline)
  const float* fc = (const float*)d_in[2];
  const float* fs = (const float*)d_in[3];
  const float* wq = (const float*)d_in[5];
  const float* wk = (const float*)d_in[6];
  const float* wv = (const float*)d_in[7];
  const float* wo = (const float*)d_in[8];
  float* out = (float*)d_out;

  char* ws = (char*)d_ws;
  size_t off = 0;
  auto grab = [&](size_t bytes) { char* p = ws + off; off += (bytes + 255) & ~(size_t)255; return p; };

  const int M  = B_ * L_;          // 4096
  const int NK = KV_ * HD_;        // 512

  _Float16* xh  = (_Float16*)grab((size_t)M * D_ * 2);
  _Float16* wqt = (_Float16*)grab((size_t)D_ * D_ * 2);
  _Float16* wkt = (_Float16*)grab((size_t)NK * D_ * 2);
  _Float16* wvt = (_Float16*)grab((size_t)NK * D_ * 2);
  _Float16* wot = (_Float16*)grab((size_t)D_ * D_ * 2);
  float*    qf  = (float*)   grab((size_t)M * D_ * 4);
  float*    kf  = (float*)   grab((size_t)M * NK * 4);
  float*    vf  = (float*)   grab((size_t)M * NK * 4);
  _Float16* qh  = (_Float16*)grab((size_t)B_ * H_  * L_ * HD_ * 2);
  _Float16* kh  = (_Float16*)grab((size_t)B_ * KV_ * L_ * HD_ * 2);
  _Float16* vth = (_Float16*)grab((size_t)B_ * KV_ * HD_ * L_ * 2);
  _Float16* aoh = (_Float16*)grab((size_t)M * D_ * 2);

  // 1) Precision conversion / weight transposes.
  {
    long n4 = (long)M * D_ / 4;
    k_cvt4<<<dim3((unsigned)((n4 + 255) / 256)), dim3(256), 0, stream>>>(x, xh, n4);
  }
  dim3 tb(32, 8);
  k_wtrans<<<dim3(D_ / 32, D_ / 32), tb, 0, stream>>>(wq, wqt, D_, D_);
  k_wtrans<<<dim3(D_ / 32, NK / 32), tb, 0, stream>>>(wk, wkt, D_, NK);
  k_wtrans<<<dim3(D_ / 32, NK / 32), tb, 0, stream>>>(wv, wvt, D_, NK);
  k_wtrans<<<dim3(D_ / 32, D_ / 32), tb, 0, stream>>>(wo, wot, D_, D_);

  // 2) QKV projections (WMMA GEMM, fp32 out).
  k_gemm<<<dim3(D_ / GBN, M / GBM), 256, 0, stream>>>(xh, wqt, qf, M, D_, D_);
  k_gemm<<<dim3(NK / GBN, M / GBM), 256, 0, stream>>>(xh, wkt, kf, M, NK, D_);
  k_gemm<<<dim3(NK / GBN, M / GBM), 256, 0, stream>>>(xh, wvt, vf, M, NK, D_);

  // 3) RoPE to head-major f16; V transposed for the PV B-fragment layout.
  {
    long nq = (long)B_ * L_ * H_  * (HD_ / 2);
    long nk = (long)B_ * L_ * KV_ * (HD_ / 2);
    k_rope<<<dim3((unsigned)((nq + 255) / 256)), 256, 0, stream>>>(qf, fc, fs, qh, H_, nq);
    k_rope<<<dim3((unsigned)((nk + 255) / 256)), 256, 0, stream>>>(kf, fc, fs, kh, KV_, nk);
  }
  k_vtrans<<<dim3(L_ / 32, HD_ / 32, B_ * KV_), tb, 0, stream>>>(vf, vth);

  // 4) Flash attention (WMMA QK^T + online softmax + WMMA PV).
  k_flash<<<dim3(L_ / 64, B_ * H_), 128, 0, stream>>>(qh, kh, vth, aoh);

  // 5) Output projection -> fp32 result.
  k_gemm<<<dim3(D_ / GBN, M / GBM), 256, 0, stream>>>(aoh, wot, out, M, D_, D_);
}
